// jambaregression_78065325572598
// MI455X (gfx1250) — compile-verified, matrix operand order checked by reading, not run on hardware
//
#include <hip/hip_runtime.h>
#include <math.h>

#define CDIV(a,b) (((a)+(b)-1)/(b))

typedef __attribute__((ext_vector_type(16))) _Float16 v16h;
typedef __attribute__((ext_vector_type(8)))  float    v8f;

// ---- model constants ----
constexpr int B_ = 8, L_ = 3072, D_ = 64, H_ = 4, DH_ = 16, GH_ = 2;
constexpr int NBF_ = 64, WIN_ = 24, NW_ = L_ / WIN_;       // 128 windows
constexpr int FF_HID_ = 256;
constexpr int D_INNER_ = 128, D_STATE_ = 16, D_CONV_ = 4, DT_RANK_ = 4;
constexpr int D_INT_ = 512, N_EXPERTS_ = 4;
constexpr int BL_ = B_ * L_;
constexpr float EPS_ = 1e-5f;
constexpr float LOG1E4 = 9.210340371976184f;

__device__ __forceinline__ float siluf(float x) { return x / (1.f + expf(-x)); }

// =====================================================================
// WMMA GEMM:  C[M,N] = A[M,K] * W[N,K]^T (+bias[N]) (+R[M,N])
// f16 inputs (converted on the fly), f32 accumulate.
// Each wave computes a 16x64 strip: A fragment loaded/converted once per
// K-step, 4 independent WMMAs issued back-to-back (register blocking).
// GUARD=false (N % 64 == 0): straight-line inner loop, no exec masking.
// GUARD=true: ragged-N handling (only used for the N=36 x_proj GEMM).
// =====================================================================
template<bool BIAS, bool RES, bool GUARD>
__global__ void k_gemm(const float* __restrict__ A, const float* __restrict__ W,
                       const float* __restrict__ bias, const float* __restrict__ R,
                       float* __restrict__ C, int M, int N, int K)
{
  int lane = threadIdx.x & 31;
  int wv   = threadIdx.x >> 5;
  int tm     = blockIdx.y * 4 + wv;        // M tile
  int tnBase = blockIdx.x * 4;             // first of 4 N tiles
  if (tm * 16 >= M) return;                // wave-uniform exit (EXEC stays full)
  int half = lane >> 4;
  int l16  = lane & 15;
  int arow = tm * 16 + l16;
  const float* ap = A + (size_t)arow * K;

  const float* wp[4];
  bool wok[4];
  #pragma unroll
  for (int nb = 0; nb < 4; ++nb) {
    int wrow = (tnBase + nb) * 16 + l16;
    wok[nb] = GUARD ? (wrow < N) : true;
    wp[nb]  = W + (size_t)(GUARD ? (wrow < N ? wrow : 0) : wrow) * K;
  }

  v8f acc[4];
  #pragma unroll
  for (int nb = 0; nb < 4; ++nb) acc[nb] = (v8f){0.f,0.f,0.f,0.f,0.f,0.f,0.f,0.f};

  for (int k0 = 0; k0 < K; k0 += 32) {
    int base = k0 + half * 8;
    v16h af;
    #pragma unroll
    for (int i = 0; i < 8; ++i) {
      af[i]     = (_Float16)ap[base + i];
      af[i + 8] = (_Float16)ap[base + 16 + i];
    }
    #pragma unroll
    for (int nb = 0; nb < 4; ++nb) {
      v16h bf;
      const float* w = wp[nb];
      #pragma unroll
      for (int i = 0; i < 8; ++i) {
        float lo = w[base + i];
        float hi = w[base + 16 + i];
        if (GUARD && !wok[nb]) { lo = 0.f; hi = 0.f; }
        bf[i]     = (_Float16)lo;
        bf[i + 8] = (_Float16)hi;
      }
      acc[nb] = __builtin_amdgcn_wmma_f32_16x16x32_f16(false, af, false, bf,
                                                       (short)0, acc[nb], false, false);
    }
  }

  #pragma unroll
  for (int nb = 0; nb < 4; ++nb) {
    int col = (tnBase + nb) * 16 + l16;
    if (!GUARD || col < N) {
      float bv = BIAS ? bias[col] : 0.f;
      #pragma unroll
      for (int r = 0; r < 8; ++r) {
        int m = tm * 16 + half * 8 + r;
        float v = acc[nb][r] + bv;
        if (RES) v += R[(size_t)m * N + col];
        C[(size_t)m * N + col] = v;
      }
    }
  }
}

// =====================================================================
// Elementwise / small kernels
// =====================================================================
__global__ void k_embed(const int* __restrict__ xtok, const float* __restrict__ tok_emb,
                        float* __restrict__ h)
{
  int idx = blockIdx.x * blockDim.x + threadIdx.x;
  if (idx >= BL_ * D_) return;
  int d = idx & 63, bl = idx >> 6, l = bl % L_;
  float pe;
  if (d < 32) { float inv = expf(-(float)d * LOG1E4 / 32.f); pe = sinf((float)l * inv); }
  else        { float inv = expf(-(float)(d - 32) * LOG1E4 / 32.f); pe = cosf((float)l * inv); }
  h[idx] = tok_emb[xtok[bl] * D_ + d] + pe;
}

__global__ void k_ln(const float* __restrict__ in, const float* __restrict__ g,
                     const float* __restrict__ b, float* __restrict__ out, int rows)
{
  int r = blockIdx.x * blockDim.x + threadIdx.x;
  if (r >= rows) return;
  const float* p = in + (size_t)r * 64;
  float s = 0.f, s2 = 0.f;
  for (int i = 0; i < 64; ++i) { float v = p[i]; s += v; s2 += v * v; }
  float mu = s * (1.f / 64.f);
  float var = s2 * (1.f / 64.f) - mu * mu;
  float inv = rsqrtf(var + EPS_);
  float* o = out + (size_t)r * 64;
  for (int i = 0; i < 64; ++i) o[i] = (p[i] - mu) * inv * g[i] + b[i];
}

__global__ void k_rms(const float* __restrict__ in, const float* __restrict__ g,
                      float* __restrict__ out, int rows)
{
  int r = blockIdx.x * blockDim.x + threadIdx.x;
  if (r >= rows) return;
  const float* p = in + (size_t)r * 64;
  float s2 = 0.f;
  for (int i = 0; i < 64; ++i) s2 += p[i] * p[i];
  float inv = rsqrtf(s2 * (1.f / 64.f) + EPS_);
  float* o = out + (size_t)r * 64;
  for (int i = 0; i < 64; ++i) o[i] = p[i] * inv * g[i];
}

__global__ void k_preshift(const float* __restrict__ t, float* __restrict__ out)
{
  int idx = blockIdx.x * blockDim.x + threadIdx.x;
  if (idx >= BL_ * 64) return;
  int d = idx & 63, bl = idx >> 6, l = bl % L_, b = bl / L_;
  float v;
  if (d < 21)       v = (l + 1 < L_) ? t[((size_t)(b * L_ + l + 1)) * 64 + d] : 0.f;
  else if (d < 42)  v = t[idx];
  else if (d < 63)  v = (l >= 1) ? t[((size_t)(b * L_ + l - 1)) * 64 + d] : 0.f;
  else              v = t[idx];
  out[idx] = v;
}

__global__ void k_rope(float* __restrict__ buf)
{
  int idx = blockIdx.x * blockDim.x + threadIdx.x;       // BL * H * 8
  if (idx >= BL_ * H_ * 8) return;
  int j = idx & 7, hh = (idx >> 3) & 3, bl = idx >> 5, l = bl % L_;
  float inv = expf(-(float)j * LOG1E4 / 8.f);
  float ang = (float)l * inv;
  float s = sinf(ang), c = cosf(ang);
  float* p = buf + (size_t)bl * 64 + hh * 16 + 2 * j;
  float x0 = p[0], x1 = p[1];
  p[0] = x0 * c - x1 * s;
  p[1] = x1 * c + x0 * s;
}

// performer feature map (query branch): per-row max
__global__ void k_perf_q(const float* __restrict__ qb, const float* __restrict__ proj,
                         float* __restrict__ qp)
{
  int row = blockIdx.x * blockDim.x + threadIdx.x;      // B*GH*L
  if (row >= B_ * GH_ * L_) return;
  int n = row % L_, bh = row / L_, hh = bh % GH_, b = bh / GH_;
  const float dn = 0.5f;           // 16^-0.25
  float qd[16]; float diag = 0.f;
  const float* qr = qb + ((size_t)(b * L_ + n)) * 64 + hh * 16;
  #pragma unroll
  for (int i = 0; i < 16; ++i) { qd[i] = qr[i] * dn; diag += qd[i] * qd[i]; }
  diag *= 0.5f;
  float dd[64]; float mx = -1e30f;
  for (int m = 0; m < 64; ++m) {
    const float* pr = proj + m * 16;
    float s = 0.f;
    #pragma unroll
    for (int i = 0; i < 16; ++i) s += qd[i] * pr[i];
    dd[m] = s; mx = fmaxf(mx, s);
  }
  float* o = qp + (size_t)row * 64;
  for (int m = 0; m < 64; ++m) o[m] = 0.125f * (expf(dd[m] - diag - mx) + 1e-4f);
}

// key branch pass1: store dd, diag, per-row max
__global__ void k_perf_k_dd(const float* __restrict__ kb, const float* __restrict__ proj,
                            float* __restrict__ kp, float* __restrict__ kdiag,
                            float* __restrict__ rowmax)
{
  int row = blockIdx.x * blockDim.x + threadIdx.x;
  if (row >= B_ * GH_ * L_) return;
  int n = row % L_, bh = row / L_, hh = bh % GH_, b = bh / GH_;
  const float dn = 0.5f;
  float kd[16]; float diag = 0.f;
  const float* kr = kb + ((size_t)(b * L_ + n)) * 64 + hh * 16;
  #pragma unroll
  for (int i = 0; i < 16; ++i) { kd[i] = kr[i] * dn; diag += kd[i] * kd[i]; }
  diag *= 0.5f;
  float mx = -1e30f;
  float* o = kp + (size_t)row * 64;
  for (int m = 0; m < 64; ++m) {
    const float* pr = proj + m * 16;
    float s = 0.f;
    #pragma unroll
    for (int i = 0; i < 16; ++i) s += kd[i] * pr[i];
    o[m] = s; mx = fmaxf(mx, s);
  }
  kdiag[row] = diag; rowmax[row] = mx;
}

__global__ void k_rowmax_reduce(const float* __restrict__ rowmax, float* __restrict__ kmax)
{
  __shared__ float sm[256];
  int bh = blockIdx.x;
  float m = -1e30f;
  for (int n = threadIdx.x; n < L_; n += 256) m = fmaxf(m, rowmax[bh * L_ + n]);
  sm[threadIdx.x] = m; __syncthreads();
  for (int s = 128; s > 0; s >>= 1) {
    if (threadIdx.x < s) sm[threadIdx.x] = fmaxf(sm[threadIdx.x], sm[threadIdx.x + s]);
    __syncthreads();
  }
  if (threadIdx.x == 0) kmax[bh] = sm[0];
}

__global__ void k_perf_k_fin(float* __restrict__ kp, const float* __restrict__ kdiag,
                             const float* __restrict__ kmax)
{
  int idx = blockIdx.x * blockDim.x + threadIdx.x;
  if (idx >= B_ * GH_ * L_ * 64) return;
  int row = idx >> 6, bh = row / L_;
  kp[idx] = 0.125f * (expf(kp[idx] - kdiag[row] - kmax[bh]) + 1e-4f);
}

__global__ void k_ksum(const float* __restrict__ kp, float* __restrict__ ksum)
{
  int tid = blockIdx.x * blockDim.x + threadIdx.x;      // B*GH*64
  if (tid >= B_ * GH_ * 64) return;
  int m = tid & 63, bh = tid >> 6;
  float s = 0.f;
  for (int n = 0; n < L_; ++n) s += kp[((size_t)(bh * L_ + n)) * 64 + m];
  ksum[tid] = s;
}

__global__ void k_ctx(const float* __restrict__ kp, const float* __restrict__ vb,
                      const unsigned char* __restrict__ mask, float* __restrict__ ctx)
{
  int tid = blockIdx.x * blockDim.x + threadIdx.x;      // B*GH*64*16
  if (tid >= B_ * GH_ * 64 * 16) return;
  int d = tid & 15, m = (tid >> 4) & 63, bh = tid >> 10;
  int hh = bh % GH_, b = bh / GH_;
  float s = 0.f;
  for (int n = 0; n < L_; ++n) {
    if (!mask[b * L_ + n]) continue;
    s += kp[((size_t)(bh * L_ + n)) * 64 + m] *
         vb[((size_t)(b * L_ + n)) * 64 + hh * 16 + d];
  }
  ctx[tid] = s;
}

__global__ void k_gout(const float* __restrict__ qp, const float* __restrict__ ksum,
                       const float* __restrict__ ctx, float* __restrict__ ao)
{
  int row = blockIdx.x * blockDim.x + threadIdx.x;      // B*GH*L
  if (row >= B_ * GH_ * L_) return;
  int n = row % L_, bh = row / L_, hh = bh % GH_, b = bh / GH_;
  float qv[64];
  const float* qr = qp + (size_t)row * 64;
  float ds = 0.f;
  const float* ks = ksum + bh * 64;
  #pragma unroll 8
  for (int m = 0; m < 64; ++m) { qv[m] = qr[m]; ds += qv[m] * ks[m]; }
  float dinv = 1.f / ds;
  float* outp = ao + ((size_t)(b * L_ + n)) * 64 + hh * 16;
  for (int d = 0; d < 16; ++d) {
    float acc = 0.f;
    #pragma unroll 8
    for (int m = 0; m < 64; ++m) acc += qv[m] * ctx[((bh * 64 + m) << 4) + d];
    outp[d] = acc * dinv;
  }
}

__global__ void k_local(const float* __restrict__ qb, const float* __restrict__ kb,
                        const float* __restrict__ vb, const unsigned char* __restrict__ mask,
                        float* __restrict__ ao)
{
  int tid = blockIdx.x * blockDim.x + threadIdx.x;      // B*2*NW*WIN
  if (tid >= B_ * 2 * NW_ * WIN_) return;
  int i = tid % WIN_, w = (tid / WIN_) % NW_, hh = (tid / (WIN_ * NW_)) & 1;
  int b = tid / (WIN_ * NW_ * 2);
  int head = GH_ + hh, n = w * WIN_ + i;
  const float scale = 0.25f;       // 16^-0.5
  float qv[16];
  const float* qr = qb + ((size_t)(b * L_ + n)) * 64 + head * 16;
  #pragma unroll
  for (int d = 0; d < 16; ++d) qv[d] = qr[d];
  // pass 1: max
  float mx = -1e30f;
  for (int x = 0; x < 3 * WIN_; ++x) {
    int kpos = (w - 1) * WIN_ + x;
    float sc = -1e9f;
    if (kpos >= 0 && kpos < L_ && mask[b * L_ + kpos]) {
      const float* kr = kb + ((size_t)(b * L_ + kpos)) * 64 + head * 16;
      float s = 0.f;
      #pragma unroll
      for (int d = 0; d < 16; ++d) s += qv[d] * kr[d];
      sc = s * scale;
    }
    mx = fmaxf(mx, sc);
  }
  // pass 2: softmax-weighted sum
  float sum = 0.f, o[16];
  #pragma unroll
  for (int d = 0; d < 16; ++d) o[d] = 0.f;
  for (int x = 0; x < 3 * WIN_; ++x) {
    int kpos = (w - 1) * WIN_ + x;
    float sc = -1e9f;
    bool ok = (kpos >= 0 && kpos < L_);
    if (ok && mask[b * L_ + kpos]) {
      const float* kr = kb + ((size_t)(b * L_ + kpos)) * 64 + head * 16;
      float s = 0.f;
      #pragma unroll
      for (int d = 0; d < 16; ++d) s += qv[d] * kr[d];
      sc = s * scale;
    }
    float pwt = expf(sc - mx);
    sum += pwt;
    if (ok) {
      const float* vr = vb + ((size_t)(b * L_ + kpos)) * 64 + head * 16;
      #pragma unroll
      for (int d = 0; d < 16; ++d) o[d] += pwt * vr[d];
    }
  }
  float inv = 1.f / sum;
  float* outp = ao + ((size_t)(b * L_ + n)) * 64 + head * 16;
  #pragma unroll
  for (int d = 0; d < 16; ++d) outp[d] = o[d] * inv;
}

__global__ void k_geglu(const float* __restrict__ big, float* __restrict__ act)
{
  int idx = blockIdx.x * blockDim.x + threadIdx.x;      // BL*256
  if (idx >= BL_ * 256) return;
  int j = idx & 255, bl = idx >> 8;
  float u = big[(size_t)bl * 512 + j];
  float g = big[(size_t)bl * 512 + 256 + j];
  float gelu = 0.5f * u * (1.f + erff(u * 0.70710678118654752f));
  act[idx] = gelu * g;
}

__global__ void k_gate_mul(const float* __restrict__ u, float* __restrict__ g, int n)
{
  int idx = blockIdx.x * blockDim.x + threadIdx.x;
  if (idx >= n) return;
  g[idx] = u[idx] * siluf(g[idx]);
}

__global__ void k_copy(const float* __restrict__ a, float* __restrict__ o, int n)
{
  int idx = blockIdx.x * blockDim.x + threadIdx.x;
  if (idx < n) o[idx] = a[idx];
}
__global__ void k_add(const float* __restrict__ a, float* __restrict__ o, int n)
{
  int idx = blockIdx.x * blockDim.x + threadIdx.x;
  if (idx < n) o[idx] += a[idx];
}

// Mamba depthwise causal conv + silu
__global__ void k_convsilu(const float* __restrict__ xz, const float* __restrict__ cw,
                           const float* __restrict__ cb, float* __restrict__ xconv)
{
  int idx = blockIdx.x * blockDim.x + threadIdx.x;      // BL*128
  if (idx >= BL_ * D_INNER_) return;
  int c = idx & 127, bl = idx >> 7, l = bl % L_, b = bl / L_;
  float acc = cb[c];
  #pragma unroll
  for (int j = 0; j < D_CONV_; ++j) {
    int lp = l + j - (D_CONV_ - 1);
    if (lp >= 0) acc += xz[((size_t)(b * L_ + lp)) * 256 + c] * cw[c * D_CONV_ + j];
  }
  xconv[idx] = siluf(acc);
}

__global__ void k_dt(const float* __restrict__ dbl, const float* __restrict__ dtw,
                     const float* __restrict__ dtb, float* __restrict__ dt)
{
  int idx = blockIdx.x * blockDim.x + threadIdx.x;      // BL*128
  if (idx >= BL_ * D_INNER_) return;
  int c = idx & 127, bl = idx >> 7;
  float s = dtb[c];
  #pragma unroll
  for (int r = 0; r < DT_RANK_; ++r) s += dbl[(size_t)bl * 36 + r] * dtw[c * DT_RANK_ + r];
  dt[idx] = (s > 20.f) ? s : log1pf(expf(s));
}

// selective scan: one block per batch, one thread per inner channel
__global__ void k_scan(const float* __restrict__ dt, const float* __restrict__ xconv,
                       const float* __restrict__ dbl, const float* __restrict__ xz,
                       const float* __restrict__ Alog, const float* __restrict__ Dp,
                       float* __restrict__ ymix)
{
  __shared__ float Bs[D_STATE_], Cs[D_STATE_];
  int b = blockIdx.x;
  int d = threadIdx.x;
  float a[D_STATE_], hst[D_STATE_];
  #pragma unroll
  for (int s = 0; s < D_STATE_; ++s) { a[s] = -expf(Alog[d * D_STATE_ + s]); hst[s] = 0.f; }
  float Dv = Dp[d];
  for (int l = 0; l < L_; ++l) {
    size_t bl = (size_t)b * L_ + l;
    if (d < D_STATE_)           Bs[d]            = dbl[bl * 36 + 4 + d];
    else if (d < 2 * D_STATE_)  Cs[d - D_STATE_] = dbl[bl * 36 + 20 + (d - D_STATE_)];
    __syncthreads();
    float dtv = dt[bl * D_INNER_ + d];
    float xv  = xconv[bl * D_INNER_ + d];
    float y = 0.f;
    #pragma unroll
    for (int s = 0; s < D_STATE_; ++s) {
      hst[s] = expf(dtv * a[s]) * hst[s] + dtv * Bs[s] * xv;
      y += hst[s] * Cs[s];
    }
    y += xv * Dv;
    float z = xz[bl * 256 + D_INNER_ + d];
    y *= siluf(z);
    ymix[bl * D_INNER_ + d] = y;
    __syncthreads();
  }
}

__global__ void k_router(const float* __restrict__ xin, const float* __restrict__ rw,
                         int* __restrict__ sel, float* __restrict__ gate)
{
  int row = blockIdx.x * blockDim.x + threadIdx.x;      // BL
  if (row >= BL_) return;
  const float* xr = xin + (size_t)row * 64;
  float lg[N_EXPERTS_]; float mx = -1e30f; int arg = 0;
  #pragma unroll
  for (int e = 0; e < N_EXPERTS_; ++e) {
    float s = 0.f;
    const float* w = rw + e * 64;
    #pragma unroll 8
    for (int i = 0; i < 64; ++i) s += xr[i] * w[i];
    lg[e] = s;
    if (s > mx) { mx = s; arg = e; }
  }
  float sum = 0.f;
  #pragma unroll
  for (int e = 0; e < N_EXPERTS_; ++e) sum += expf(lg[e] - mx);
  sel[row] = arg;
  gate[row] = 1.f / sum;           // softmax prob of the argmax
}

__global__ void k_moe_mid(const float* __restrict__ xin, const float* __restrict__ fc1,
                          const int* __restrict__ sel, float* __restrict__ act)
{
  int idx = blockIdx.x * blockDim.x + threadIdx.x;      // BL*512
  if (idx >= BL_ * D_INT_) return;
  int j = idx & 511, bl = idx >> 9;
  int e = sel[bl];
  const float* xr = xin + (size_t)bl * 64;
  const float* wu = fc1 + ((size_t)e * 1024 + j) * 64;
  const float* wg = fc1 + ((size_t)e * 1024 + 512 + j) * 64;
  float u = 0.f, g = 0.f;
  #pragma unroll 8
  for (int i = 0; i < 64; ++i) { u += xr[i] * wu[i]; g += xr[i] * wg[i]; }
  act[idx] = u * siluf(g);
}

__global__ void k_moe_out(const float* __restrict__ act, const float* __restrict__ fc2,
                          const int* __restrict__ sel, const float* __restrict__ gate,
                          float* __restrict__ h)
{
  int idx = blockIdx.x * blockDim.x + threadIdx.x;      // BL*64
  if (idx >= BL_ * 64) return;
  int d = idx & 63, bl = idx >> 6;
  int e = sel[bl];
  const float* ar = act + (size_t)bl * D_INT_;
  const float* w = fc2 + ((size_t)e * 64 + d) * D_INT_;
  float s = 0.f;
  #pragma unroll 8
  for (int i = 0; i < D_INT_; ++i) s += ar[i] * w[i];
  h[idx] = gate[bl] * s;
}

__global__ void k_maskmul(float* __restrict__ h, const unsigned char* __restrict__ mask)
{
  int idx = blockIdx.x * blockDim.x + threadIdx.x;
  if (idx >= BL_ * 64) return;
  if (!mask[idx >> 6]) h[idx] = 0.f;
}

// conv tower: kernel 3, stride 2, pad 1. inBLD: input is [b, t, c]; else [b, c, t]
__global__ void k_convt(const float* __restrict__ in, const float* __restrict__ w,
                        const float* __restrict__ bias, float* __restrict__ out,
                        int Cin, int Cout, int Lin, int Lout, int inBLD)
{
  int idx = blockIdx.x * blockDim.x + threadIdx.x;
  int total = B_ * Cout * Lout;
  if (idx >= total) return;
  int t = idx % Lout, oc = (idx / Lout) % Cout, b = idx / (Lout * Cout);
  float acc = bias[oc];
  for (int ic = 0; ic < Cin; ++ic) {
    #pragma unroll
    for (int j = 0; j < 3; ++j) {
      int s = 2 * t + j - 1;
      if (s >= 0 && s < Lin) {
        float xv = inBLD ? in[((size_t)(b * Lin + s)) * Cin + ic]
                         : in[((size_t)(b * Cin + ic)) * Lin + s];
        acc += w[(oc * Cin + ic) * 3 + j] * xv;
      }
    }
  }
  out[((size_t)(b * Cout) + oc) * Lout + t] = acc;
}

__global__ void k_head(const float* __restrict__ c3, const float* __restrict__ tpm,
                       const int* __restrict__ cell, const float* __restrict__ cell_emb,
                       const float* __restrict__ fc1w, const float* __restrict__ fc1b,
                       const float* __restrict__ fc2w, const float* __restrict__ fc2b,
                       float* __restrict__ out)
{
  int b = threadIdx.x;
  if (b >= B_) return;
  float feat[17];
  for (int c = 0; c < 4; ++c) {
    float s = 0.f, m = -1e30f;
    for (int t = 0; t < 384; ++t) {
      float v = c3[((size_t)(b * 4) + c) * 384 + t];
      s += v; m = fmaxf(m, v);
    }
    feat[c] = s / 384.f; feat[4 + c] = m;
  }
  feat[8] = tpm[b];
  int ct = cell[b];
  for (int i = 0; i < 8; ++i) feat[9 + i] = cell_emb[ct * 8 + i];
  float y[24];
  for (int j = 0; j < 24; ++j) {
    float s = fc1b[j];
    #pragma unroll
    for (int i = 0; i < 17; ++i) s += fc1w[j * 17 + i] * feat[i];
    y[j] = siluf(s);
  }
  float o = fc2b[0];
  for (int j = 0; j < 24; ++j) o += fc2w[j] * y[j];
  out[b] = o;
}

// =====================================================================
// Host orchestration
// =====================================================================
extern "C" void kernel_launch(void* const* d_in, const int* in_sizes, int n_in,
                              void* d_out, int out_size, void* d_ws, size_t ws_size,
                              hipStream_t stream)
{
  (void)in_sizes; (void)n_in; (void)out_size; (void)ws_size;
  int p = 0;
  auto F = [&]() -> const float* { return (const float*)d_in[p++]; };

  const int*           xtok = (const int*)d_in[p++];
  const unsigned char* mask = (const unsigned char*)d_in[p++];
  const float*         tpm  = F();
  const int*           cell = (const int*)d_in[p++];
  const float*      tok_emb = F();

  struct PL { const float *ln_g,*ln_b,*q_w,*k_w,*v_w,*out_w,*out_b,*proj,
                          *fln_g,*fln_b,*w1_w,*w1_b,*w2_w,*w2_b; } pl[2];
  for (int i = 0; i < 2; ++i) {
    pl[i].ln_g=F(); pl[i].ln_b=F(); pl[i].q_w=F(); pl[i].k_w=F(); pl[i].v_w=F();
    pl[i].out_w=F(); pl[i].out_b=F(); pl[i].proj=F();
    pl[i].fln_g=F(); pl[i].fln_b=F(); pl[i].w1_w=F(); pl[i].w1_b=F();
    pl[i].w2_w=F(); pl[i].w2_b=F();
  }
  const float* final_g = F();
  const float* final_b = F();

  struct MB { const float *n1,*n2,*inp,*cw,*cb,*xp,*dtw,*dtb,*Alog,*Dp,*outp,
                          *router,*fc1,*fc2; bool moe; } mb[3];
  for (int i = 0; i < 3; ++i) {
    mb[i].n1=F(); mb[i].n2=F(); mb[i].inp=F(); mb[i].cw=F(); mb[i].cb=F();
    mb[i].xp=F(); mb[i].dtw=F(); mb[i].dtb=F(); mb[i].Alog=F(); mb[i].Dp=F();
    mb[i].outp=F();
    if (i % 2 == 0) { mb[i].moe=false; mb[i].router=nullptr; mb[i].fc1=F(); mb[i].fc2=F(); }
    else            { mb[i].moe=true;  mb[i].router=F();     mb[i].fc1=F(); mb[i].fc2=F(); }
  }
  const float* c1w=F(); const float* c1b=F();
  const float* c2w=F(); const float* c2b=F();
  const float* c3w=F(); const float* c3b=F();
  const float* cemb=F();
  const float* hf1w=F(); const float* hf1b=F();
  const float* hf2w=F(); const float* hf2b=F();

  // ---- workspace ----
  char* wsb = (char*)d_ws;
  size_t off = 0;
  auto alloc = [&](size_t nfloats) -> float* {
    float* ptr = (float*)(wsb + off);
    off += nfloats * sizeof(float);
    off = (off + 255) & ~(size_t)255;
    return ptr;
  };
  float* hbuf  = alloc((size_t)BL_ * 64);
  float* res   = alloc((size_t)BL_ * 64);
  float* t0    = alloc((size_t)BL_ * 64);
  float* xin   = alloc((size_t)BL_ * 64);
  float* qb    = alloc((size_t)BL_ * 64);
  float* kb    = alloc((size_t)BL_ * 64);
  float* vb    = alloc((size_t)BL_ * 64);
  float* ao    = alloc((size_t)BL_ * 64);
  float* qp    = alloc((size_t)B_ * GH_ * L_ * 64);
  float* kp    = alloc((size_t)B_ * GH_ * L_ * 64);
  float* kdiag = alloc((size_t)B_ * GH_ * L_);
  float* rowmx = alloc((size_t)B_ * GH_ * L_);
  float* kmax  = alloc(64);
  float* ksum  = alloc((size_t)B_ * GH_ * 64);
  float* ctx   = alloc((size_t)B_ * GH_ * 64 * 16);
  float* big   = alloc((size_t)BL_ * 512);
  float* act   = alloc((size_t)BL_ * 512);
  float* xz    = alloc((size_t)BL_ * 256);
  float* xconv = alloc((size_t)BL_ * 128);
  float* dtb2  = alloc((size_t)BL_ * 128);
  float* ymix  = alloc((size_t)BL_ * 128);
  float* dbl   = alloc((size_t)BL_ * 36);
  int*   selb  = (int*)alloc((size_t)BL_);
  float* gateb = alloc((size_t)BL_);
  float* cv1   = alloc((size_t)B_ * 16 * 1536);
  float* cv2   = alloc((size_t)B_ * 8 * 768);
  float* cv3   = alloc((size_t)B_ * 4 * 384);

  auto gemm = [&](const float* A, const float* W, const float* bias, const float* R,
                  float* C, int M, int N, int K, bool hasBias, bool hasRes) {
    dim3 grid(CDIV(N, 64), CDIV(CDIV(M, 16), 4));
    dim3 blk(128);
    if (N % 64) {
      // ragged N (only the x_proj GEMM, no bias / no residual)
      k_gemm<false,false,true ><<<grid, blk, 0, stream>>>(A,W,bias,R,C,M,N,K);
    } else if (hasBias && hasRes) {
      k_gemm<true ,true ,false><<<grid, blk, 0, stream>>>(A,W,bias,R,C,M,N,K);
    } else if (hasBias) {
      k_gemm<true ,false,false><<<grid, blk, 0, stream>>>(A,W,bias,R,C,M,N,K);
    } else if (hasRes) {
      k_gemm<false,true ,false><<<grid, blk, 0, stream>>>(A,W,bias,R,C,M,N,K);
    } else {
      k_gemm<false,false,false><<<grid, blk, 0, stream>>>(A,W,bias,R,C,M,N,K);
    }
  };
  #define EW(kern, n, ...) kern<<<CDIV((n), 256), 256, 0, stream>>>(__VA_ARGS__)

  // ---- embedding ----
  EW(k_embed, BL_ * 64, xtok, tok_emb, hbuf);

  // ---- performer layers ----
  for (int li = 0; li < 2; ++li) {
    const PL& a = pl[li];
    EW(k_ln, BL_, hbuf, a.ln_g, a.ln_b, t0, BL_);
    EW(k_preshift, BL_ * 64, t0, xin);
    gemm(xin, a.q_w, nullptr, nullptr, qb, BL_, 64, 64, false, false);
    gemm(xin, a.k_w, nullptr, nullptr, kb, BL_, 64, 64, false, false);
    gemm(xin, a.v_w, nullptr, nullptr, vb, BL_, 64, 64, false, false);
    EW(k_rope, BL_ * H_ * 8, qb);
    EW(k_rope, BL_ * H_ * 8, kb);
    EW(k_perf_q, B_ * GH_ * L_, qb, a.proj, qp);
    EW(k_perf_k_dd, B_ * GH_ * L_, kb, a.proj, kp, kdiag, rowmx);
    k_rowmax_reduce<<<B_ * GH_, 256, 0, stream>>>(rowmx, kmax);
    EW(k_perf_k_fin, B_ * GH_ * L_ * 64, kp, kdiag, kmax);
    EW(k_ksum, B_ * GH_ * 64, kp, ksum);
    EW(k_ctx, B_ * GH_ * 64 * 16, kp, vb, mask, ctx);
    EW(k_gout, B_ * GH_ * L_, qp, ksum, ctx, ao);
    EW(k_local, B_ * 2 * NW_ * WIN_, qb, kb, vb, mask, ao);
    gemm(ao, a.out_w, a.out_b, hbuf, hbuf, BL_, 64, 64, true, true);
    // feed-forward (GEGLU)
    EW(k_ln, BL_, hbuf, a.fln_g, a.fln_b, t0, BL_);
    EW(k_preshift, BL_ * 64, t0, xin);
    gemm(xin, a.w1_w, a.w1_b, nullptr, big, BL_, 512, 64, true, false);
    EW(k_geglu, BL_ * 256, big, act);
    gemm(act, a.w2_w, a.w2_b, hbuf, hbuf, BL_, 64, 256, true, true);
  }

  // ---- final layer norm (in place) ----
  EW(k_ln, BL_, hbuf, final_g, final_b, hbuf, BL_);

  // ---- mamba blocks ----
  for (int i = 0; i < 3; ++i) {
    const MB& m = mb[i];
    if (i == 0) EW(k_copy, BL_ * 64, hbuf, res, BL_ * 64);
    else        EW(k_add,  BL_ * 64, hbuf, res, BL_ * 64);
    EW(k_rms, BL_, res, m.n1, xin, BL_);
    // mixer
    gemm(xin, m.inp, nullptr, nullptr, xz, BL_, 256, 64, false, false);
    EW(k_convsilu, BL_ * 128, xz, m.cw, m.cb, xconv);
    gemm(xconv, m.xp, nullptr, nullptr, dbl, BL_, 36, 128, false, false);
    EW(k_dt, BL_ * 128, dbl, m.dtw, m.dtb, dtb2);
    k_scan<<<B_, D_INNER_, 0, stream>>>(dtb2, xconv, dbl, xz, m.Alog, m.Dp, ymix);
    gemm(ymix, m.outp, nullptr, nullptr, t0, BL_, 64, 128, false, false);
    EW(k_add, BL_ * 64, t0, res, BL_ * 64);      // residual = hs + residual
    EW(k_rms, BL_, res, m.n2, xin, BL_);
    if (!m.moe) {
      gemm(xin, m.fc1,            nullptr, nullptr, big, BL_, 512, 64, false, false); // u
      gemm(xin, m.fc1 + 512 * 64, nullptr, nullptr, act, BL_, 512, 64, false, false); // g
      EW(k_gate_mul, BL_ * 512, big, act, BL_ * 512);                                 // act = u*silu(g)
      gemm(act, m.fc2, nullptr, nullptr, hbuf, BL_, 64, 512, false, false);
    } else {
      EW(k_router, BL_, xin, m.router, selb, gateb);
      EW(k_moe_mid, BL_ * 512, xin, m.fc1, selb, act);
      EW(k_moe_out, BL_ * 64, act, m.fc2, selb, gateb, hbuf);
    }
  }

  // ---- mask + conv tower + head ----
  EW(k_maskmul, BL_ * 64, hbuf, mask);
  EW(k_convt, B_ * 16 * 1536, hbuf, c1w, c1b, cv1, 64, 16, 3072, 1536, 1);
  EW(k_convt, B_ * 8 * 768,   cv1,  c2w, c2b, cv2, 16, 8, 1536, 768, 0);
  EW(k_convt, B_ * 4 * 384,   cv2,  c3w, c3b, cv3, 8, 4, 768, 384, 0);
  k_head<<<1, 32, 0, stream>>>(cv3, tpm, cell, cemb, hf1w, hf1b, hf2w, hf2b, (float*)d_out);
  #undef EW
}
